// SWEGNN_Adforce_61211873902756
// MI455X (gfx1250) — matrix operand order repaired and checked by reading, not compiled, not run on hardware
//
#include <hip/hip_runtime.h>
#include <hip/hip_bf16.h>

typedef __attribute__((ext_vector_type(16))) _Float16 v16h;
typedef __attribute__((ext_vector_type(8)))  _Float16 v8h;
typedef __attribute__((ext_vector_type(4)))  _Float16 v4h;
typedef __attribute__((ext_vector_type(8)))  float    v8f;

union F16x16 { v16h v; v8h h[2]; };

// ---------------------------------------------------------------------------
// Packed A-fragment index (ISA 7.12.2, 16-bit A 16x32 layout).
// Element (row i, col k) of an [Mpad, Kdim] f16 matrix lives at:
//   tile (mt = i/16, kt = k/32), lane = (i%16) + half*16, byte j in lane's 32B.
//   k' = k%32:  k'<16 -> half=k'/8, j=k'%8 ;  k'>=16 -> half=(k'-16)/8, j=8+(k'%8)
// A wave then fetches one whole fragment as 32 contiguous bytes per lane.
// ---------------------------------------------------------------------------
__device__ __forceinline__ size_t pa_index(int i, int k, int KT) {
    int mt = i >> 4, r = i & 15;
    int kt = k >> 5, kk = k & 31;
    int half, j;
    if (kk < 16) { half = kk >> 3; j = kk & 7; }
    else         { half = (kk - 16) >> 3; j = 8 + (kk & 7); }
    int lane = r + (half << 4);
    return (((size_t)mt * KT + kt) * 32 + lane) * 16 + j;
}

// ---------------------------------------------------------------------------
// Weight repack: W [Kdim, Nout] f32 row-major -> packed f16 B-fragments.
// Lane L of fragment (nt,kt): column n = nt*16 + (L&15), elements 0..7 =
// K kt*32 + (L>>4)*8 + 0..7, elements 8..15 = K kt*32 + 16 + (L>>4)*8 + 0..7.
// ---------------------------------------------------------------------------
__global__ void pack_w_kernel(const float* __restrict__ W, _Float16* __restrict__ P,
                              int Kdim, int Nout) {
    int idx = blockIdx.x * blockDim.x + threadIdx.x;
    int KT = Kdim >> 5;
    int total = (Nout >> 4) * KT * 32;
    if (idx >= total) return;
    int lane = idx & 31;
    int tile = idx >> 5;
    int kt = tile % KT;
    int nt = tile / KT;
    int n  = (nt << 4) + (lane & 15);
    int k0 = (kt << 5) + ((lane >> 4) << 3);
    _Float16* o = P + (size_t)idx * 16;
#pragma unroll
    for (int j = 0; j < 16; ++j)
        o[j] = (_Float16)W[(size_t)(k0 + j + ((j >> 3) << 3)) * Nout + n];
}

// ---------------------------------------------------------------------------
// Layer-1 encoders (tiny K): H = relu(X @ W1 + b1), written directly in the
// packed A-fragment f16 layout (rows >= M zero-filled so WMMA tiles are clean).
// ---------------------------------------------------------------------------
__global__ void enc1_pack_kernel(const float* __restrict__ X, const float* __restrict__ W,
                                 const float* __restrict__ b, _Float16* __restrict__ P,
                                 int M, int Mpad, int Kin) {
    int idx = blockIdx.x * blockDim.x + threadIdx.x;
    if (idx >= Mpad * 256) return;
    int i = idx >> 8;
    int j = idx & 255;
    float acc = 0.f;
    if (i < M) {
        acc = b[j];
        const float* xr = X + (size_t)i * Kin;
        for (int k = 0; k < Kin; ++k)
            acc += xr[k] * W[k * 256 + j];
        acc = fmaxf(acc, 0.f);
    }
    P[pa_index(i, j, 8)] = (_Float16)acc;   // KT = 256/32
}

// ---------------------------------------------------------------------------
// Shared epilogue: OUTMODE 0 = f32 row-major, 1 = f16 packed-A (feeds the next
// WMMA GEMM), 2 = f16 row-major (feeds gather loader / elementwise kernels).
// ---------------------------------------------------------------------------
template<bool RELU, int OUTMODE>
__device__ __forceinline__ void gemm_epilogue(
    v8f acc[2][2], const float* __restrict__ bias, void* __restrict__ C,
    int M, int Nout, int m0, int n0, int wm, int wn, int lane, int half) {
#pragma unroll
    for (int ntl = 0; ntl < 2; ++ntl) {
        int n = n0 + (wn * 2 + ntl) * 16 + (lane & 15);
        float bv = bias[n];
#pragma unroll
        for (int mt = 0; mt < 2; ++mt) {
            int mb = m0 + wm * 32 + mt * 16 + half * 8;
#pragma unroll
            for (int r = 0; r < 8; ++r) {
                int m = mb + r;
                float val = acc[mt][ntl][r] + bv;
                if (RELU) val = fmaxf(val, 0.f);
                if (OUTMODE == 0) {
                    if (m < M) ((float*)C)[(size_t)m * Nout + n] = val;
                } else if (OUTMODE == 1) {
                    ((_Float16*)C)[pa_index(m, n, Nout >> 5)] = (_Float16)val;
                } else {
                    if (m < M) ((_Float16*)C)[(size_t)m * Nout + n] = (_Float16)val;
                }
            }
        }
    }
}

// ---------------------------------------------------------------------------
// WMMA GEMM, A pre-packed f16 fragments: no LDS, no barriers, no conversions.
// Block 256 thr = 8 waves (4M x 2N), tile 128x64, K step 32.
// Grid.x covers Mpad/128 (packed A has zero-filled tail rows).
// ---------------------------------------------------------------------------
template<bool RELU, int OUTMODE>
__global__ __launch_bounds__(256) void gemm_packA_kernel(
    const _Float16* __restrict__ Ap, const _Float16* __restrict__ Wp,
    const float* __restrict__ bias, void* __restrict__ C,
    int M, int Kdim, int Nout)
{
    const int t    = threadIdx.x;
    const int lane = t & 31;
    const int w    = t >> 5;
    const int wm   = w & 3;
    const int wn   = w >> 2;
    const int m0   = blockIdx.x * 128;
    const int n0   = blockIdx.y * 64;
    const int KT   = Kdim >> 5;
    const int half = lane >> 4;

    v8f acc[2][2] = {};

    for (int kt = 0; kt < KT; ++kt) {
        F16x16 af[2];
#pragma unroll
        for (int mt = 0; mt < 2; ++mt) {
            int mtile = (m0 >> 4) + wm * 2 + mt;
            af[mt].v = *(const v16h*)(Ap + (((size_t)mtile * KT + kt) * 32 + lane) * 16);
        }
#pragma unroll
        for (int ntl = 0; ntl < 2; ++ntl) {
            int nt = (n0 >> 4) + wn * 2 + ntl;
            F16x16 bf;
            bf.v = *(const v16h*)(Wp + (((size_t)nt * KT + kt) * 32 + lane) * 16);
#pragma unroll
            for (int mt = 0; mt < 2; ++mt) {
                acc[mt][ntl] = __builtin_amdgcn_wmma_f32_16x16x32_f16(
                    false, af[mt].v, false, bf.v, (short)0, acc[mt][ntl],
                    false, false);
            }
        }
    }
    gemm_epilogue<RELU, OUTMODE>(acc, bias, C, M, Nout, m0, n0, wm, wn, lane, half);
}

// ---------------------------------------------------------------------------
// WMMA GEMM, gather-concat A (g layer 1): row i = [e[i], xs[src[i]], xs[dst[i]]],
// all f16 row-major; Kdim = 768. xs (25 MB) stays L2-resident on the 192MB L2.
// ---------------------------------------------------------------------------
template<bool RELU, int OUTMODE>
__global__ __launch_bounds__(256) void gemm_gather_kernel(
    const _Float16* __restrict__ Ae, const _Float16* __restrict__ Xs,
    const int* __restrict__ srcIdx, const int* __restrict__ dstIdx,
    const _Float16* __restrict__ Wp, const float* __restrict__ bias,
    void* __restrict__ C, int M, int Kdim, int Nout)
{
    __shared__ _Float16 lA[128 * 40];   // f16 tile, padded stride

    const int t    = threadIdx.x;
    const int lane = t & 31;
    const int w    = t >> 5;
    const int wm   = w & 3;
    const int wn   = w >> 2;
    const int m0   = blockIdx.x * 128;
    const int n0   = blockIdx.y * 64;
    const int KT   = Kdim >> 5;
    const int half = lane >> 4;

    v8f acc[2][2] = {};

    for (int kt = 0; kt < KT; ++kt) {
        __syncthreads();
        // stage 128x32 f16 tile: 512 x 16B segments, 2 per thread
#pragma unroll
        for (int i = 0; i < 2; ++i) {
            int s   = t + i * 256;
            int row = s >> 2;
            int c8  = (s & 3) << 3;
            int gr  = m0 + row;
            v8h v;
#pragma unroll
            for (int q = 0; q < 8; ++q) v[q] = (_Float16)0.f;
            if (gr < M) {
                int kg = (kt << 5) + c8;
                const _Float16* rp;
                if (kg < 256)      rp = Ae + (size_t)gr * 256 + kg;
                else if (kg < 512) rp = Xs + (size_t)srcIdx[gr] * 256 + (kg - 256);
                else               rp = Xs + (size_t)dstIdx[gr] * 256 + (kg - 512);
                v = *(const v8h*)rp;
            }
            *(v8h*)&lA[row * 40 + c8] = v;
        }
        __syncthreads();

        F16x16 af[2];
#pragma unroll
        for (int mt = 0; mt < 2; ++mt) {
            int m = wm * 32 + mt * 16 + (lane & 15);
            const _Float16* lr = &lA[m * 40];
            af[mt].h[0] = *(const v8h*)(lr + half * 8);
            af[mt].h[1] = *(const v8h*)(lr + 16 + half * 8);
        }
#pragma unroll
        for (int ntl = 0; ntl < 2; ++ntl) {
            int nt = (n0 >> 4) + wn * 2 + ntl;
            F16x16 bf;
            bf.v = *(const v16h*)(Wp + (((size_t)nt * KT + kt) * 32 + lane) * 16);
#pragma unroll
            for (int mt = 0; mt < 2; ++mt) {
                acc[mt][ntl] = __builtin_amdgcn_wmma_f32_16x16x32_f16(
                    false, af[mt].v, false, bf.v, (short)0, acc[mt][ntl],
                    false, false);
            }
        }
    }
    gemm_epilogue<RELU, OUTMODE>(acc, bias, C, M, Nout, m0, n0, wm, wn, lane, half);
}

// ---------------------------------------------------------------------------
// Decoder layer-1 GEMM: A = tanh(h) with h f32 row-major; f32->f16 conversion
// fused into the LDS staging. Output f32 row-major, relu.
// ---------------------------------------------------------------------------
__global__ __launch_bounds__(256) void gemm_dec1_kernel(
    const float* __restrict__ A, const _Float16* __restrict__ Wp,
    const float* __restrict__ bias, float* __restrict__ C,
    int M, int Kdim, int Nout)
{
    __shared__ _Float16 lA[128 * 40];

    const int t    = threadIdx.x;
    const int lane = t & 31;
    const int w    = t >> 5;
    const int wm   = w & 3;
    const int wn   = w >> 2;
    const int m0   = blockIdx.x * 128;
    const int n0   = blockIdx.y * 64;
    const int KT   = Kdim >> 5;
    const int half = lane >> 4;

    v8f acc[2][2] = {};

    for (int kt = 0; kt < KT; ++kt) {
        __syncthreads();
#pragma unroll
        for (int i = 0; i < 4; ++i) {
            int s   = t + i * 256;
            int row = s >> 3;
            int c4  = (s & 7) << 2;
            int gr  = m0 + row;
            float4 v = make_float4(0.f, 0.f, 0.f, 0.f);
            if (gr < M)
                v = *(const float4*)(A + (size_t)gr * Kdim + (kt << 5) + c4);
            v4h hv;
            hv[0] = (_Float16)tanhf(v.x); hv[1] = (_Float16)tanhf(v.y);
            hv[2] = (_Float16)tanhf(v.z); hv[3] = (_Float16)tanhf(v.w);
            *(v4h*)&lA[row * 40 + c4] = hv;
        }
        __syncthreads();

        F16x16 af[2];
#pragma unroll
        for (int mt = 0; mt < 2; ++mt) {
            int m = wm * 32 + mt * 16 + (lane & 15);
            const _Float16* lr = &lA[m * 40];
            af[mt].h[0] = *(const v8h*)(lr + half * 8);
            af[mt].h[1] = *(const v8h*)(lr + 16 + half * 8);
        }
#pragma unroll
        for (int ntl = 0; ntl < 2; ++ntl) {
            int nt = (n0 >> 4) + wn * 2 + ntl;
            F16x16 bf;
            bf.v = *(const v16h*)(Wp + (((size_t)nt * KT + kt) * 32 + lane) * 16);
#pragma unroll
            for (int mt = 0; mt < 2; ++mt) {
                acc[mt][ntl] = __builtin_amdgcn_wmma_f32_16x16x32_f16(
                    false, af[mt].v, false, bf.v, (short)0, acc[mt][ntl],
                    false, false);
            }
        }
    }
    gemm_epilogue<true, 0>(acc, bias, C, M, Nout, m0, n0, wm, wn, lane, half);
}

// ---------------------------------------------------------------------------
__global__ void fillz_kernel(float* __restrict__ p, int n) {
    int idx = blockIdx.x * blockDim.x + threadIdx.x;
    if (idx < n) p[idx] = 0.f;
}

__global__ void deg_kernel(const int* __restrict__ dst, float* __restrict__ deg, int E) {
    int idx = blockIdx.x * blockDim.x + threadIdx.x;
    if (idx < E) atomicAdd(&deg[dst[idx]], 1.0f);
}

__global__ void invdeg_kernel(float* __restrict__ deg, int N) {
    int idx = blockIdx.x * blockDim.x + threadIdx.x;
    if (idx < N) deg[idx] = 1.0f / fmaxf(deg[idx], 1.0f);
}

// m = g * (h[src] - h[dst]); aggr[dst] += m.  g is f16 (GEMM-precision data),
// h/aggr stay f32 for the residual accumulation.  64 threads/edge.
__global__ void mp_edge_kernel(const _Float16* __restrict__ g, const float* __restrict__ h,
                               const int* __restrict__ src, const int* __restrict__ dst,
                               float* __restrict__ aggr, int E) {
    int idx = blockIdx.x * blockDim.x + threadIdx.x;
    if (idx >= E * 64) return;
    int e  = idx >> 6;
    int c4 = (idx & 63) << 2;
    int s = src[e], d = dst[e];
    v4h gv = *(const v4h*)(g + (size_t)e * 256 + c4);
    float4 hs = *(const float4*)(h + (size_t)s * 256 + c4);
    float4 hd = *(const float4*)(h + (size_t)d * 256 + c4);
    float* ap = aggr + (size_t)d * 256 + c4;
    atomicAdd(ap + 0, (float)gv[0] * (hs.x - hd.x));
    atomicAdd(ap + 1, (float)gv[1] * (hs.y - hd.y));
    atomicAdd(ap + 2, (float)gv[2] * (hs.z - hd.z));
    atomicAdd(ap + 3, (float)gv[3] * (hs.w - hd.w));
}

__global__ void mp_update_kernel(float* __restrict__ h, const float* __restrict__ aggr,
                                 const float* __restrict__ inv, int N) {
    int idx = blockIdx.x * blockDim.x + threadIdx.x;
    if (idx >= N * 256) return;
    h[idx] += aggr[idx] * inv[idx >> 8];
}

__global__ void dec2_kernel(const float* __restrict__ Hd, const float* __restrict__ W,
                            const float* __restrict__ b, float* __restrict__ out, int N) {
    int idx = blockIdx.x * blockDim.x + threadIdx.x;
    if (idx >= N * 3) return;
    int i = idx / 3, o = idx % 3;
    float acc = b[o];
    const float* hr = Hd + (size_t)i * 256;
    for (int k = 0; k < 256; ++k)
        acc += hr[k] * W[k * 3 + o];
    out[idx] = acc;
}

// ---------------------------------------------------------------------------
extern "C" void kernel_launch(void* const* d_in, const int* in_sizes, int n_in,
                              void* d_out, int out_size, void* d_ws, size_t ws_size,
                              hipStream_t stream) {
    const float* statf = (const float*)d_in[0];
    const float* dynf  = (const float*)d_in[1];
    const int*   eidx  = (const int*)  d_in[2];
    const float* eattr = (const float*)d_in[3];
    const float* sW1 = (const float*)d_in[4];  const float* sb1 = (const float*)d_in[5];
    const float* sW2 = (const float*)d_in[6];  const float* sb2 = (const float*)d_in[7];
    const float* dW1 = (const float*)d_in[8];  const float* db1 = (const float*)d_in[9];
    const float* dW2 = (const float*)d_in[10]; const float* db2 = (const float*)d_in[11];
    const float* eW1 = (const float*)d_in[12]; const float* eb1 = (const float*)d_in[13];
    const float* eW2 = (const float*)d_in[14]; const float* eb2 = (const float*)d_in[15];
    const float* gW1 = (const float*)d_in[16]; const float* gb1 = (const float*)d_in[17];
    const float* gW2 = (const float*)d_in[18]; const float* gb2 = (const float*)d_in[19];
    const float* oW1 = (const float*)d_in[20]; const float* ob1 = (const float*)d_in[21];
    const float* oW2 = (const float*)d_in[22]; const float* ob2 = (const float*)d_in[23];

    const int N = in_sizes[0] / 10;
    const int E = in_sizes[2] / 2;
    const int Npad = (N + 127) & ~127;
    const int Epad = (E + 127) & ~127;
    const int* src = eidx;
    const int* dst = eidx + E;

    // ---- workspace carve-out -------------------------------------------------
    char* ws = (char*)d_ws;
    size_t off = 0;
    auto carve = [&](size_t bytes) -> void* {
        void* p = ws + off;
        off += (bytes + 255) & ~(size_t)255;
        return p;
    };
    _Float16* scrP = (_Float16*)carve((size_t)Epad * 256 * 2); // packed h1 / g1 / (f32 hdec)
    _Float16* bufE = (_Float16*)carve((size_t)Epad * 256 * 2); // e f16, then g f16
    _Float16* xs16 = (_Float16*)carve((size_t)N * 256 * 2);    // x_s f16 (L2-resident)
    float*    hbuf = (float*)   carve((size_t)N * 256 * 4);    // x_d -> h (f32)
    float*    aggr = (float*)   carve((size_t)N * 256 * 4);
    float*    invd = (float*)   carve((size_t)N * 4);
    _Float16* pWs  = (_Float16*)carve((size_t)256 * 256 * 2);
    _Float16* pWd  = (_Float16*)carve((size_t)256 * 256 * 2);
    _Float16* pWe  = (_Float16*)carve((size_t)256 * 256 * 2);
    _Float16* pWg1 = (_Float16*)carve((size_t)768 * 256 * 2);
    _Float16* pWg2 = (_Float16*)carve((size_t)256 * 256 * 2);
    _Float16* pWo1 = (_Float16*)carve((size_t)256 * 256 * 2);
    (void)ws_size; (void)n_in; (void)out_size;

    auto pack = [&](const float* W, _Float16* P, int Kd, int No) {
        int total = (No / 16) * (Kd / 32) * 32;
        pack_w_kernel<<<(total + 255) / 256, 256, 0, stream>>>(W, P, Kd, No);
    };
    pack(sW2, pWs, 256, 256);
    pack(dW2, pWd, 256, 256);
    pack(eW2, pWe, 256, 256);
    pack(gW1, pWg1, 768, 256);
    pack(gW2, pWg2, 256, 256);
    pack(oW1, pWo1, 256, 256);

    dim3 blk(256);
    auto ggrid = [&](int Mp) { return dim3(Mp / 128, 256 / 64); };

    // ---- encoders (layer1 -> packed f16 A, layer2 = packed WMMA GEMM) -------
    enc1_pack_kernel<<<((size_t)Npad * 256 + 255) / 256, blk, 0, stream>>>(
        statf, sW1, sb1, scrP, N, Npad, 10);
    gemm_packA_kernel<false, 2><<<ggrid(Npad), blk, 0, stream>>>(
        scrP, pWs, sb2, xs16, N, 256, 256);                    // x_s f16

    enc1_pack_kernel<<<((size_t)Npad * 256 + 255) / 256, blk, 0, stream>>>(
        dynf, dW1, db1, scrP, N, Npad, 12);
    gemm_packA_kernel<false, 0><<<ggrid(Npad), blk, 0, stream>>>(
        scrP, pWd, db2, hbuf, N, 256, 256);                    // h = x_d f32

    enc1_pack_kernel<<<((size_t)Epad * 256 + 255) / 256, blk, 0, stream>>>(
        eattr, eW1, eb1, scrP, E, Epad, 3);
    gemm_packA_kernel<false, 2><<<ggrid(Epad), blk, 0, stream>>>(
        scrP, pWe, eb2, bufE, E, 256, 256);                    // e f16

    // ---- edge-conditioned weights g -----------------------------------------
    gemm_gather_kernel<true, 1><<<ggrid(Epad), blk, 0, stream>>>(
        bufE, xs16, src, dst, pWg1, gb1, scrP, E, 768, 256);   // g1 packed f16
    gemm_packA_kernel<false, 2><<<ggrid(Epad), blk, 0, stream>>>(
        scrP, pWg2, gb2, bufE, E, 256, 256);                   // g f16 (over e)

    // ---- degree normalization ------------------------------------------------
    fillz_kernel<<<(N + 255) / 256, blk, 0, stream>>>(invd, N);
    deg_kernel<<<(E + 255) / 256, blk, 0, stream>>>(dst, invd, E);
    invdeg_kernel<<<(N + 255) / 256, blk, 0, stream>>>(invd, N);

    // ---- K = 2 hops of SWE message passing ----------------------------------
    for (int hop = 0; hop < 2; ++hop) {
        fillz_kernel<<<((size_t)N * 256 + 255) / 256, blk, 0, stream>>>(aggr, N * 256);
        mp_edge_kernel<<<((size_t)E * 64 + 255) / 256, blk, 0, stream>>>(
            bufE, hbuf, src, dst, aggr, E);
        mp_update_kernel<<<((size_t)N * 256 + 255) / 256, blk, 0, stream>>>(
            hbuf, aggr, invd, N);
    }

    // ---- decoder -------------------------------------------------------------
    float* hdec = (float*)scrP;   // reuse scratch (needs N*256*4 <= Epad*256*2)
    gemm_dec1_kernel<<<ggrid(Npad), blk, 0, stream>>>(
        hbuf, pWo1, ob1, hdec, N, 256, 256);
    dec2_kernel<<<((size_t)N * 3 + 255) / 256, blk, 0, stream>>>(
        hdec, oW2, ob2, (float*)d_out, N);
}
